// Attention_45191645889170
// MI455X (gfx1250) — matrix operand order repaired and checked by reading, not compiled, not run on hardware
//
#include <hip/hip_runtime.h>

// ---------------------------------------------------------------------------
// Multi-head attention (b=8, n=1024, dim=1024, heads=16, head_dim=64)
// bf16 WMMA pipeline for gfx1250 (wave32, v_wmma_f32_16x16x32_bf16)
// ---------------------------------------------------------------------------

typedef __bf16 bf16_t;
typedef __attribute__((ext_vector_type(16))) bf16_t v16bf;
typedef __attribute__((ext_vector_type(8)))  float  v8f;

#define DIM      1024
#define HEADS    16
#define HEAD_DIM 64
#define SEQ      1024
#define BATCH    8
#define SCALE    0.125f   // HEAD_DIM^-0.5

union Frag {
    v16bf v;
    uint4 q[2];
    unsigned short s[16];
};

__device__ __forceinline__ unsigned short f2bf(float f) {
    union { float f; unsigned int u; } c; c.f = f;
    unsigned int u = c.u;
    unsigned int r = u + 0x7FFFu + ((u >> 16) & 1u);   // round-to-nearest-even
    return (unsigned short)(r >> 16);
}

// ------------------------- fp32 -> bf16 conversion (x4 vectorized) ---------
__global__ void __launch_bounds__(256) cvt_f32_bf16(const float4* __restrict__ in,
                                                    ushort4* __restrict__ out,
                                                    int n4) {
    int i = blockIdx.x * 256 + threadIdx.x;
    if (i < n4) {
        float4 f = in[i];
        ushort4 o;
        o.x = f2bf(f.x); o.y = f2bf(f.y); o.z = f2bf(f.z); o.w = f2bf(f.w);
        out[i] = o;
    }
}

// ------------------------- shared GEMM wave-tile core ----------------------
// acc[2][4] = A[m0:m0+32, :] @ B[n0:n0+64, :]^T   (A row-major [M,K], B
// row-major [N,K] like an nn.Linear weight).  One wave computes a 32x64 slab:
// each B fragment is reused by two A fragments -> 8 WMMA per 6 loads.
__device__ __forceinline__ void gemm_wave_tile32(const unsigned short* __restrict__ A,
                                                 const unsigned short* __restrict__ B,
                                                 int m0, int n0, int K, v8f acc[2][4]) {
    const int lane = threadIdx.x & 31;
    const int half = lane >> 4;
    const int lm   = lane & 15;
    const unsigned short* arow0 = A + (size_t)(m0 + lm) * K + half * 8;
    const unsigned short* arow1 = A + (size_t)(m0 + 16 + lm) * K + half * 8;
    for (int k = 0; k < K; k += 32) {
        Frag a0, a1;
        a0.q[0] = *(const uint4*)(arow0 + k);
        a0.q[1] = *(const uint4*)(arow0 + k + 16);
        a1.q[0] = *(const uint4*)(arow1 + k);
        a1.q[1] = *(const uint4*)(arow1 + k + 16);
#pragma unroll
        for (int t = 0; t < 4; ++t) {
            const unsigned short* brow =
                B + (size_t)(n0 + t * 16 + lm) * K + k + half * 8;
            Frag bf;
            bf.q[0] = *(const uint4*)(brow);
            bf.q[1] = *(const uint4*)(brow + 16);
            acc[0][t] = __builtin_amdgcn_wmma_f32_16x16x32_bf16(
                false, a0.v, false, bf.v, (short)0, acc[0][t], false, false);
            acc[1][t] = __builtin_amdgcn_wmma_f32_16x16x32_bf16(
                false, a1.v, false, bf.v, (short)0, acc[1][t], false, false);
        }
    }
}

// ------------------------- QKV projection ---------------------------------
// xb [8192,1024] bf16, wb = W_qkv [3072,1024] bf16.
// Scatter into Q [b,h,n,d] (pre-scaled), K [b,h,n,d], Vt [b,h,d,n].
__global__ void __launch_bounds__(128) qkv_gemm(const unsigned short* __restrict__ xb,
                                                const unsigned short* __restrict__ wb,
                                                unsigned short* __restrict__ Qd,
                                                unsigned short* __restrict__ Kd,
                                                unsigned short* __restrict__ Vt) {
    const int wave = threadIdx.x >> 5;
    const int lane = threadIdx.x & 31;
    const int half = lane >> 4;
    const int lm   = lane & 15;
    const int m0 = blockIdx.y * 128 + wave * 32;
    const int n0 = blockIdx.x * 64;

    v8f acc[2][4] = {};
    gemm_wave_tile32(xb, wb, m0, n0, DIM, acc);

    const int sel  = n0 / DIM;                 // 0=Q 1=K 2=V
    const int head = (n0 % DIM) / HEAD_DIM;    // 64-wide slab -> single head
#pragma unroll
    for (int g = 0; g < 2; ++g) {
#pragma unroll
        for (int t = 0; t < 4; ++t) {
#pragma unroll
            for (int r = 0; r < 8; ++r) {
                const int m = m0 + g * 16 + r + half * 8;
                const int b = m >> 10;
                const int i = m & (SEQ - 1);
                const int dd = t * 16 + lm;
                const size_t bh = (size_t)(b * HEADS + head);
                const float val = acc[g][t][r];
                if (sel == 0)
                    Qd[(bh * SEQ + i) * HEAD_DIM + dd] = f2bf(val * SCALE);
                else if (sel == 1)
                    Kd[(bh * SEQ + i) * HEAD_DIM + dd] = f2bf(val);
                else
                    Vt[(bh * HEAD_DIM + dd) * SEQ + i] = f2bf(val);
            }
        }
    }
}

// ------------------------- flash attention ---------------------------------
// One wave owns a 16-row Q tile of one (b,h); streams K/V in 32-column blocks
// with online softmax.  Output written bf16 as [b, i, h*64+d].
__global__ void __launch_bounds__(128) attn_kernel(const unsigned short* __restrict__ Qd,
                                                   const unsigned short* __restrict__ Kd,
                                                   const unsigned short* __restrict__ Vt,
                                                   unsigned short* __restrict__ O) {
    __shared__ __align__(16) unsigned short plds[4][16 * 32];

    const int wave = threadIdx.x >> 5;
    const int lane = threadIdx.x & 31;
    const int half = lane >> 4;
    const int lm   = lane & 15;

    const int bh = blockIdx.x >> 4;             // 0..127  (b*16 + h)
    const int tg = blockIdx.x & 15;
    const int i0 = (tg * 4 + wave) * 16;

    const unsigned short* Qb = Qd + (size_t)bh * SEQ * HEAD_DIM;
    const unsigned short* Kb = Kd + (size_t)bh * SEQ * HEAD_DIM;
    const unsigned short* Vb = Vt + (size_t)bh * HEAD_DIM * SEQ;

    // Q tile A-fragments for the two 32-wide chunks of head_dim
    Frag qa[2];
    {
        const unsigned short* qrow = Qb + (size_t)(i0 + lm) * HEAD_DIM + half * 8;
        qa[0].q[0] = *(const uint4*)(qrow);
        qa[0].q[1] = *(const uint4*)(qrow + 16);
        qa[1].q[0] = *(const uint4*)(qrow + 32);
        qa[1].q[1] = *(const uint4*)(qrow + 48);
    }

    v8f o[4] = {};
    float rmax[8], rsum[8];
#pragma unroll
    for (int r = 0; r < 8; ++r) { rmax[r] = -__builtin_inff(); rsum[r] = 0.0f; }

    unsigned short* pl = plds[wave];

    for (int jc = 0; jc < SEQ; jc += 32) {
        // ---- scores S = Q @ K^T for columns [jc, jc+32) : two 16x16 tiles
        v8f s[2];
#pragma unroll
        for (int t = 0; t < 2; ++t) {
            v8f c = {};
#pragma unroll
            for (int kc = 0; kc < 2; ++kc) {
                const unsigned short* krow =
                    Kb + (size_t)(jc + t * 16 + lm) * HEAD_DIM + kc * 32 + half * 8;
                Frag kf;
                kf.q[0] = *(const uint4*)(krow);
                kf.q[1] = *(const uint4*)(krow + 16);
                c = __builtin_amdgcn_wmma_f32_16x16x32_bf16(
                    false, qa[kc].v, false, kf.v, (short)0, c, false, false);
            }
            s[t] = c;
        }

        // ---- online softmax (rows live across 16-lane halves)
#pragma unroll
        for (int r = 0; r < 8; ++r) {
            float mloc = fmaxf(s[0][r], s[1][r]);
#pragma unroll
            for (int off = 8; off >= 1; off >>= 1)
                mloc = fmaxf(mloc, __shfl_xor(mloc, off, 32));
            const float mnew  = fmaxf(rmax[r], mloc);
            const float alpha = __expf(rmax[r] - mnew);
            rmax[r] = mnew;
            const float p0 = __expf(s[0][r] - mnew);
            const float p1 = __expf(s[1][r] - mnew);
            s[0][r] = p0;
            s[1][r] = p1;
            float ls = p0 + p1;
#pragma unroll
            for (int off = 8; off >= 1; off >>= 1)
                ls += __shfl_xor(ls, off, 32);
            rsum[r] = rsum[r] * alpha + ls;
#pragma unroll
            for (int t = 0; t < 4; ++t) o[t][r] *= alpha;
        }

        // ---- transpose P (C-layout) -> A-layout via per-wave LDS staging
#pragma unroll
        for (int r = 0; r < 8; ++r) {
            const int m = r + half * 8;
            pl[m * 32 + lm]      = f2bf(s[0][r]);
            pl[m * 32 + 16 + lm] = f2bf(s[1][r]);
        }
        Frag pa;
        {
            const unsigned short* prow = pl + lm * 32 + half * 8;
            pa.q[0] = *(const uint4*)(prow);
            pa.q[1] = *(const uint4*)(prow + 16);
        }

        // ---- O += P @ V  (V transposed layout: contiguous B-fragment loads)
#pragma unroll
        for (int t = 0; t < 4; ++t) {
            const unsigned short* vrow =
                Vb + (size_t)(t * 16 + lm) * SEQ + jc + half * 8;
            Frag vf;
            vf.q[0] = *(const uint4*)(vrow);
            vf.q[1] = *(const uint4*)(vrow + 16);
            o[t] = __builtin_amdgcn_wmma_f32_16x16x32_bf16(
                false, pa.v, false, vf.v, (short)0, o[t], false, false);
        }
    }

    // ---- normalize and store bf16 [b, i, h*64 + d]
    const int b = bh >> 4;
    const int h = bh & 15;
#pragma unroll
    for (int r = 0; r < 8; ++r) {
        const float inv = 1.0f / rsum[r];
        const int m = i0 + r + half * 8;
        const size_t base = ((size_t)b * SEQ + m) * DIM + h * HEAD_DIM;
#pragma unroll
        for (int t = 0; t < 4; ++t)
            O[base + t * 16 + lm] = f2bf(o[t][r] * inv);
    }
}

// ------------------------- output projection -------------------------------
// out[8192,1024] f32 = attn[8192,1024]bf16 @ W_proj[1024,1024]^T + b_proj
__global__ void __launch_bounds__(128) proj_gemm(const unsigned short* __restrict__ ab,
                                                 const unsigned short* __restrict__ wp,
                                                 const float* __restrict__ bias,
                                                 float* __restrict__ out) {
    const int wave = threadIdx.x >> 5;
    const int lane = threadIdx.x & 31;
    const int half = lane >> 4;
    const int lm   = lane & 15;
    const int m0 = blockIdx.y * 128 + wave * 32;
    const int n0 = blockIdx.x * 64;

    v8f acc[2][4] = {};
    gemm_wave_tile32(ab, wp, m0, n0, DIM, acc);

#pragma unroll
    for (int t = 0; t < 4; ++t) {
        const int n = n0 + t * 16 + lm;
        const float bs = bias[n];
#pragma unroll
        for (int g = 0; g < 2; ++g) {
#pragma unroll
            for (int r = 0; r < 8; ++r) {
                const int m = m0 + g * 16 + r + half * 8;
                out[(size_t)m * DIM + n] = acc[g][t][r] + bs;
            }
        }
    }
}

// ---------------------------------------------------------------------------
extern "C" void kernel_launch(void* const* d_in, const int* in_sizes, int n_in,
                              void* d_out, int out_size, void* d_ws, size_t ws_size,
                              hipStream_t stream) {
    const float* x     = (const float*)d_in[0];   // [8,1024,1024]
    const float* Wqkv  = (const float*)d_in[1];   // [3072,1024]
    const float* Wproj = (const float*)d_in[2];   // [1024,1024]
    const float* bproj = (const float*)d_in[3];   // [1024]
    float* out = (float*)d_out;

    const int NX  = BATCH * SEQ * DIM;        // 8388608
    const int NWQ = 3 * DIM * DIM;            // 3145728
    const int NWP = DIM * DIM;                // 1048576
    const int NBH = BATCH * HEADS * SEQ * HEAD_DIM;  // 8388608

    unsigned short* ws = (unsigned short*)d_ws;
    size_t off = 0;
    unsigned short* xb    = ws + off; off += (size_t)NX;
    unsigned short* wqkvb = ws + off; off += (size_t)NWQ;
    unsigned short* wprojb= ws + off; off += (size_t)NWP;
    unsigned short* Qd    = ws + off; off += (size_t)NBH;
    unsigned short* Kd    = ws + off; off += (size_t)NBH;
    unsigned short* Vt    = ws + off; off += (size_t)NBH;
    unsigned short* attnb = ws + off; off += (size_t)NX;
    (void)ws_size; (void)in_sizes; (void)n_in; (void)out_size;

    cvt_f32_bf16<<<(NX / 4 + 255) / 256, 256, 0, stream>>>(
        (const float4*)x, (ushort4*)xb, NX / 4);
    cvt_f32_bf16<<<(NWQ / 4 + 255) / 256, 256, 0, stream>>>(
        (const float4*)Wqkv, (ushort4*)wqkvb, NWQ / 4);
    cvt_f32_bf16<<<(NWP / 4 + 255) / 256, 256, 0, stream>>>(
        (const float4*)Wproj, (ushort4*)wprojb, NWP / 4);

    qkv_gemm<<<dim3(3 * DIM / 64, BATCH * SEQ / 128), 128, 0, stream>>>(
        xb, wqkvb, Qd, Kd, Vt);

    attn_kernel<<<BATCH * HEADS * (SEQ / 16) / 4, 128, 0, stream>>>(
        Qd, Kd, Vt, attnb);

    proj_gemm<<<dim3(DIM / 64, BATCH * SEQ / 128), 128, 0, stream>>>(
        attnb, wprojb, bproj, out);
}
